// MultiHeadAttention_1065151889592
// MI455X (gfx1250) — matrix operand order repaired and checked by reading, not compiled, not run on hardware
//
#include <hip/hip_runtime.h>
#include <hip/hip_bf16.h>

// ---------------- types ----------------
typedef __bf16 bf16_t;
typedef __attribute__((ext_vector_type(16))) __bf16 v16bf;
typedef __attribute__((ext_vector_type(8)))  float  v8f;

static constexpr int B_  = 2;
static constexpr int N_  = 2048;
static constexpr int D_  = 2048;
static constexpr int H_  = 16;
static constexpr int DK_ = 128;
static constexpr size_t S1 = (size_t)B_ * N_ * D_;   // 8,388,608 elements
static constexpr size_t DD = (size_t)D_ * D_;        // 4,194,304 elements

__device__ inline v8f wmma_bf16(v16bf a, v16bf b, v8f c) {
  return __builtin_amdgcn_wmma_f32_16x16x32_bf16(false, a, false, b, (short)0, c,
                                                 false, false);
}

// ---- CDNA5 async global->LDS copy (no data VGPRs, tracked by ASYNCcnt) ----
// Generic LDS addresses hold the LDS byte offset in their low 32 bits
// (ISA 10.2); vdst operand = LDS address VGPR.
__device__ __forceinline__ void async_copy16(const void* gaddr, void* lds) {
  unsigned l32 = (unsigned)(unsigned long long)lds;
  asm volatile("global_load_async_to_lds_b128 %0, %1, off"
               :: "v"(l32), "v"(gaddr) : "memory");
}
__device__ __forceinline__ void async_wait0() {
  asm volatile("s_wait_asynccnt 0x0" ::: "memory");
}

// A fragment (16x32, bf16): lane l holds row m=l&15, k in [kb,kb+16),
// kb = (l>>4)*16. 'row' points at the k-slice base for this fragment's row.
__device__ inline v16bf load_frag_a(const bf16_t* row, int kb) {
  v16bf a;
#pragma unroll
  for (int i = 0; i < 16; ++i) {
    a[i] = row[kb + i];
  }
  return a;
}

// B fragment (32x16, bf16): lane l holds col n=l&15, 16 contiguous k starting at
// (l>>4)*16 — pass pointer to those 16 contiguous halfs.
__device__ inline v16bf load_frag_b(const bf16_t* p) {
  v16bf b;
#pragma unroll
  for (int i = 0; i < 16; ++i) b[i] = p[i];
  return b;
}

// ---------------- f32 -> bf16 bulk convert (8 elems / thread) ----------------------
__global__ void __launch_bounds__(256)
conv_f32_bf16(const float* __restrict__ src, bf16_t* __restrict__ dst, int count) {
  int idx = blockIdx.x * blockDim.x + threadIdx.x;
  int base = idx << 3;
  if (base >= count) return;
  float4 v0 = *(const float4*)(src + base);
  float4 v1 = *(const float4*)(src + base + 4);
  union { uint4 u; bf16_t h[8]; } pk;
  pk.h[0] = (bf16_t)v0.x; pk.h[1] = (bf16_t)v0.y;
  pk.h[2] = (bf16_t)v0.z; pk.h[3] = (bf16_t)v0.w;
  pk.h[4] = (bf16_t)v1.x; pk.h[5] = (bf16_t)v1.y;
  pk.h[6] = (bf16_t)v1.z; pk.h[7] = (bf16_t)v1.w;
  *(uint4*)(dst + base) = pk.u;
}

// ---------------- GEMM: C[M,Ncol] = A[M,K] * W[Ncol,K]^T (bf16 in, f32 out) --------
// 128x128 block tile, TK=32, 4 waves, each wave 64x64 (16 WMMAs/k-step),
// double-buffered LDS filled by async global->LDS copies (no staging VGPRs).
#define TK 32

__global__ void __launch_bounds__(128)
gemm_bf16_nt(const bf16_t* __restrict__ A, const bf16_t* __restrict__ W,
             float* __restrict__ C, int M, int Ncol, int K) {
  __shared__ __align__(16) bf16_t As[2][128][TK];   // 2 x 8 KB
  __shared__ __align__(16) bf16_t Bs[2][128][TK];   // 2 x 8 KB

  const int tid  = threadIdx.x;
  const int lane = tid & 31;
  const int wave = tid >> 5;
  const int wr   = wave & 1;    // 2 row-groups of 64 rows
  const int wc   = wave >> 1;   // 2 col-groups of 64 cols
  const int bm   = blockIdx.y * 128;
  const int bn   = blockIdx.x * 128;

  v8f zero = {};
  v8f acc[4][4];
#pragma unroll
  for (int mi = 0; mi < 4; ++mi)
#pragma unroll
    for (int ni = 0; ni < 4; ++ni) acc[mi][ni] = zero;

  // staging map: tile = 128x32 bf16 = 512 16-byte chunks; 4 chunks / thread
  // chunk c: row = c>>2, col = (c&3)*8
  // ---- prologue: async-fill buffer 0 ----
#pragma unroll
  for (int i = 0; i < 4; ++i) {
    int c = tid + i * 128;
    int row = c >> 2, col = (c & 3) << 3;
    async_copy16(A + (size_t)(bm + row) * K + col, &As[0][row][col]);
    async_copy16(W + (size_t)(bn + row) * K + col, &Bs[0][row][col]);
  }
  async_wait0();
  __syncthreads();

  int buf = 0;
  for (int k0 = 0; k0 < K; k0 += TK, buf ^= 1) {
    const bool next = (k0 + TK) < K;
    if (next) {   // async copies for the next tile run under the WMMA block
#pragma unroll
      for (int i = 0; i < 4; ++i) {
        int c = tid + i * 128;
        int row = c >> 2, col = (c & 3) << 3;
        async_copy16(A + (size_t)(bm + row) * K + k0 + TK + col,
                     &As[buf ^ 1][row][col]);
        async_copy16(W + (size_t)(bn + row) * K + k0 + TK + col,
                     &Bs[buf ^ 1][row][col]);
      }
    }

    // compute on current buffer: 16 WMMAs
    v16bf af[4];
#pragma unroll
    for (int mi = 0; mi < 4; ++mi)
      af[mi] = load_frag_a(&As[buf][64 * wr + 16 * mi + (lane & 15)][0],
                           (lane >> 4) << 4);
#pragma unroll
    for (int ni = 0; ni < 4; ++ni) {
      v16bf bf = load_frag_b(&Bs[buf][64 * wc + 16 * ni + (lane & 15)][(lane >> 4) << 4]);
#pragma unroll
      for (int mi = 0; mi < 4; ++mi)
        acc[mi][ni] = wmma_bf16(af[mi], bf, acc[mi][ni]);
    }

    async_wait0();     // next-tile LDS writes landed
    __syncthreads();
  }

  // epilogue: C layout — VGPR r: m = r + (lane>>4)*8, n = lane&15
#pragma unroll
  for (int mi = 0; mi < 4; ++mi)
#pragma unroll
    for (int ni = 0; ni < 4; ++ni)
#pragma unroll
      for (int r = 0; r < 8; ++r) {
        int m = bm + 64 * wr + 16 * mi + r + ((lane >> 4) << 3);
        int n = bn + 64 * wc + 16 * ni + (lane & 15);
        C[(size_t)m * Ncol + n] = acc[mi][ni][r];
      }
}

// ---------------- RoPE + pack: src f32 [B,N,H*dk] -> dst bf16 [B,H,N,dk] -----------
__global__ void __launch_bounds__(256)
rope_pack(const float* __restrict__ src, bf16_t* __restrict__ dstb,
          float* __restrict__ dstf /* nullable */) {
  int idx = blockIdx.x * blockDim.x + threadIdx.x;   // over B*H*N*(dk/2)
  const int total = B_ * H_ * N_ * (DK_ / 2);
  if (idx >= total) return;
  int j = idx & 63;
  int t = idx >> 6;
  int n = t % N_; t /= N_;
  int h = t % H_;
  int b = t / H_;

  const float* p = src + ((size_t)(b * N_ + n)) * D_ + h * DK_ + 2 * j;
  float x1 = p[0], x2 = p[1];
  float ang = (float)n * __expf(-(float)j * (9.210340371976184f / 64.0f));
  float s, c;
  __sincosf(ang, &s, &c);
  float o1 = x1 * c - x2 * s;
  float o2 = x1 * s + x2 * c;
  size_t o = ((size_t)((b * H_ + h) * N_ + n)) * DK_ + 2 * j;
  dstb[o]     = (bf16_t)o1;
  dstb[o + 1] = (bf16_t)o2;
  if (dstf) { dstf[o] = o1; dstf[o + 1] = o2; }
}

// ---------------- V pack: f32 [B,N,H*dk] -> f32 + bf16 [B,H,N,dk] ------------------
__global__ void __launch_bounds__(256)
v_pack(const float* __restrict__ src, bf16_t* __restrict__ dstb,
       float* __restrict__ dstf) {
  int idx = blockIdx.x * blockDim.x + threadIdx.x;   // over B*H*N*dk
  const int total = B_ * H_ * N_ * DK_;
  if (idx >= total) return;
  int dd = idx & (DK_ - 1);
  int t  = idx >> 7;
  int n  = t % N_; t /= N_;
  int h  = t % H_;
  int b  = t / H_;
  float v = src[((size_t)(b * N_ + n)) * D_ + h * DK_ + dd];
  size_t o = ((size_t)((b * H_ + h) * N_ + n)) * DK_ + dd;
  dstf[o] = v;
  dstb[o] = (bf16_t)v;
}

// ---------------- Flash attention (causal, online softmax) ------------------------
#define FA_BM 64
#define FA_BN 64

__global__ void __launch_bounds__(128)
flash_attn(const bf16_t* __restrict__ Qb, const bf16_t* __restrict__ Kb,
           const bf16_t* __restrict__ Vb,
           bf16_t* __restrict__ Ob /* bf16 [B,N,D] for the output GEMM */) {
  __shared__ __align__(16) bf16_t Qs[FA_BM][DK_];     // 16 KB
  __shared__ __align__(16) bf16_t Ks[FA_BN][DK_];     // 16 KB
  __shared__ __align__(16) bf16_t Vt[DK_][FA_BN];     // 16 KB, transposed (d-major)
  __shared__ __align__(16) bf16_t Ps[4][16][FA_BN];   // 8 KB per-wave P tiles

  const int tid  = threadIdx.x;
  const int lane = tid & 31;
  const int w    = tid >> 5;          // 4 waves, 16 rows each
  const int q0   = blockIdx.x * FA_BM;
  const int h    = blockIdx.y;
  const int b    = blockIdx.z;
  const size_t headoff = ((size_t)(b * H_ + h)) * N_ * DK_;

  // async-load Q block (bf16, already roped); waited together with first K tile
  {
    const uint4* Qg = (const uint4*)(Qb + headoff + (size_t)q0 * DK_);
    uint4* Ql = (uint4*)&Qs[0][0];
    for (int i = tid; i < FA_BM * DK_ / 8; i += 128) async_copy16(Qg + i, Ql + i);
  }

  v8f zero = {};
  v8f oacc[8];
#pragma unroll
  for (int d = 0; d < 8; ++d) oacc[d] = zero;
  float mrow[8], lrow[8];
#pragma unroll
  for (int r = 0; r < 8; ++r) { mrow[r] = -1e30f; lrow[r] = 0.0f; }

  const float sc = 0.08838834764831845f;  // 1/sqrt(128)
  const int kend = q0 + FA_BM;            // causal upper bound

  for (int kb = 0; kb < kend; kb += FA_BN) {
    __syncthreads();
    // async-stage K block
    {
      const uint4* Kg = (const uint4*)(Kb + headoff + (size_t)kb * DK_);
      uint4* Kl = (uint4*)&Ks[0][0];
      for (int i = tid; i < FA_BN * DK_ / 8; i += 128) async_copy16(Kg + i, Kl + i);
    }
    // stage V transposed: Vt[d][kseq]; coalesced global reads, packed b32 stores
    {
      const bf16_t* Vg = Vb + headoff + (size_t)kb * DK_;
#pragma unroll 4
      for (int ks = 0; ks < FA_BN; ks += 2) {
        union { unsigned int u; bf16_t h[2]; } pk;
        pk.h[0] = Vg[(size_t)ks * DK_ + tid];
        pk.h[1] = Vg[(size_t)(ks + 1) * DK_ + tid];
        *(unsigned int*)&Vt[tid][ks] = pk.u;
      }
    }
    async_wait0();
    __syncthreads();

    // S = Q * K^T  (dk=128 -> 4 WMMA k-steps)
    v8f s[4];
#pragma unroll
    for (int nt = 0; nt < 4; ++nt) s[nt] = zero;
#pragma unroll
    for (int kt = 0; kt < 4; ++kt) {
      v16bf a = load_frag_a(&Qs[16 * w + (lane & 15)][32 * kt], (lane >> 4) << 4);
#pragma unroll
      for (int nt = 0; nt < 4; ++nt) {
        v16bf bf = load_frag_b(&Ks[16 * nt + (lane & 15)][32 * kt + ((lane >> 4) << 4)]);
        s[nt] = wmma_bf16(a, bf, s[nt]);
      }
    }

    // scale + causal mask + online softmax (16-lane group reductions, wave32)
#pragma unroll
    for (int r = 0; r < 8; ++r) {
      int m = r + ((lane >> 4) << 3);
      int q = q0 + 16 * w + m;
      float vmax = -1e30f;
#pragma unroll
      for (int nt = 0; nt < 4; ++nt) {
        int kk = kb + 16 * nt + (lane & 15);
        float val = s[nt][r] * sc;
        if (kk > q) val = -1e30f;
        s[nt][r] = val;
        vmax = fmaxf(vmax, val);
      }
#pragma unroll
      for (int off = 1; off < 16; off <<= 1)
        vmax = fmaxf(vmax, __shfl_xor(vmax, off, 32));
      float mnew  = fmaxf(mrow[r], vmax);
      float alpha = __expf(mrow[r] - mnew);
      float rsum = 0.0f;
#pragma unroll
      for (int nt = 0; nt < 4; ++nt) {
        float p = __expf(s[nt][r] - mnew);
        s[nt][r] = p;
        rsum += p;
      }
#pragma unroll
      for (int off = 1; off < 16; off <<= 1) rsum += __shfl_xor(rsum, off, 32);
      lrow[r] = lrow[r] * alpha + rsum;
      mrow[r] = mnew;
#pragma unroll
      for (int d = 0; d < 8; ++d) oacc[d][r] = oacc[d][r] * alpha;
    }

    // P -> LDS (bf16, per-wave region), then O += P * V
#pragma unroll
    for (int r = 0; r < 8; ++r) {
      int m = r + ((lane >> 4) << 3);
#pragma unroll
      for (int nt = 0; nt < 4; ++nt)
        Ps[w][m][16 * nt + (lane & 15)] = (bf16_t)s[nt][r];
    }
    __syncthreads();   // make P visible across lanes (cross-lane LDS dependency)

#pragma unroll
    for (int kt = 0; kt < 2; ++kt) {
      v16bf a = load_frag_a(&Ps[w][lane & 15][32 * kt], (lane >> 4) << 4);
#pragma unroll
      for (int dt = 0; dt < 8; ++dt) {
        v16bf bf = load_frag_b(&Vt[16 * dt + (lane & 15)][32 * kt + ((lane >> 4) << 4)]);
        oacc[dt] = wmma_bf16(a, bf, oacc[dt]);
      }
    }
  }

  // epilogue: normalized O written bf16 in [B,N,H*dk] layout for the output GEMM
#pragma unroll
  for (int r = 0; r < 8; ++r) {
    float inv = 1.0f / lrow[r];
    int m = r + ((lane >> 4) << 3);
    int nrow = q0 + 16 * w + m;
#pragma unroll
    for (int dt = 0; dt < 8; ++dt) {
      int col = h * DK_ + 16 * dt + (lane & 15);
      Ob[((size_t)(b * N_ + nrow)) * D_ + col] = (bf16_t)(oacc[dt][r] * inv);
    }
  }
}

// ---------------- launcher ----------------
extern "C" void kernel_launch(void* const* d_in, const int* in_sizes, int n_in,
                              void* d_out, int out_size, void* d_ws, size_t ws_size,
                              hipStream_t stream) {
  (void)in_sizes; (void)n_in; (void)out_size; (void)ws_size;
  const float* x  = (const float*)d_in[0];
  const float* wq = (const float*)d_in[1];
  const float* wk = (const float*)d_in[2];
  const float* wv = (const float*)d_in[3];
  const float* wo = (const float*)d_in[4];

  float* out  = (float*)d_out;        // [B,N,D]
  float* Kout = out + S1;             // [B,H,N,dk] (roped)
  float* Vout = out + 2 * S1;         // [B,H,N,dk]

  float*  Qf  = (float*)d_ws;         // f32 projections [B,N,D]
  float*  Kf  = Qf + S1;
  float*  Vf  = Kf + S1;
  bf16_t* Qb  = (bf16_t*)(Vf + S1);   // bf16 [B,H,N,dk]
  bf16_t* Kb  = Qb + S1;
  bf16_t* Vb  = Kb + S1;
  bf16_t* Ob  = Vb + S1;              // bf16 attention output [B,N,D]
  bf16_t* Xb  = Ob + S1;              // bf16 input [B,N,D]
  bf16_t* Wqb = Xb + S1;              // bf16 weights [D,D]
  bf16_t* Wkb = Wqb + DD;
  bf16_t* Wvb = Wkb + DD;
  bf16_t* Wob = Wvb + DD;

  // one-time f32 -> bf16 conversions (removes converts from GEMM hot loops)
  conv_f32_bf16<<<(int)(S1 / 8 + 255) / 256, 256, 0, stream>>>(x,  Xb,  (int)S1);
  conv_f32_bf16<<<(int)(DD / 8 + 255) / 256, 256, 0, stream>>>(wq, Wqb, (int)DD);
  conv_f32_bf16<<<(int)(DD / 8 + 255) / 256, 256, 0, stream>>>(wk, Wkb, (int)DD);
  conv_f32_bf16<<<(int)(DD / 8 + 255) / 256, 256, 0, stream>>>(wv, Wvb, (int)DD);
  conv_f32_bf16<<<(int)(DD / 8 + 255) / 256, 256, 0, stream>>>(wo, Wob, (int)DD);

  const int M = B_ * N_;              // 4096
  dim3 gg(D_ / 128, M / 128);         // (16, 32)

  gemm_bf16_nt<<<gg, 128, 0, stream>>>(Xb, Wqb, Qf, M, D_, D_);
  gemm_bf16_nt<<<gg, 128, 0, stream>>>(Xb, Wkb, Kf, M, D_, D_);
  gemm_bf16_nt<<<gg, 128, 0, stream>>>(Xb, Wvb, Vf, M, D_, D_);

  {
    int total = B_ * H_ * N_ * (DK_ / 2);
    rope_pack<<<(total + 255) / 256, 256, 0, stream>>>(Qf, Qb, nullptr);
    rope_pack<<<(total + 255) / 256, 256, 0, stream>>>(Kf, Kb, Kout);
  }
  {
    int total = B_ * H_ * N_ * DK_;
    v_pack<<<(total + 255) / 256, 256, 0, stream>>>(Vf, Vb, Vout);
  }

  flash_attn<<<dim3(N_ / FA_BM, H_, B_), 128, 0, stream>>>(Qb, Kb, Vb, Ob);

  gemm_bf16_nt<<<gg, 128, 0, stream>>>(Ob, Wob, out, M, D_, D_);
}
